// GraphMixer_Grad_73744588472663
// MI455X (gfx1250) — matrix operand order repaired and checked by reading, not compiled, not run on hardware
//
#include <hip/hip_runtime.h>
#include <hip/hip_bf16.h>

// Problem constants (match reference)
#define NODES 50000
#define EDGES 800000
#define FIN   512
#define DD    128
#define NBLK  3

typedef __attribute__((ext_vector_type(16))) __bf16 v16bf;
typedef __attribute__((ext_vector_type(8)))  __bf16 v8bf;
typedef __attribute__((ext_vector_type(8)))  float  v8f;

// ---------- small device helpers ----------
__device__ __forceinline__ unsigned short f2b(float f) {
  unsigned u = __float_as_uint(f);
  u += 0x7FFFu + ((u >> 16) & 1u);            // round-to-nearest-even
  return (unsigned short)(u >> 16);
}
__device__ __forceinline__ unsigned encf(float f) {   // order-preserving float->uint
  unsigned u = __float_as_uint(f);
  return (u & 0x80000000u) ? ~u : (u | 0x80000000u);
}
__device__ __forceinline__ float decf(unsigned u) {
  unsigned v = (u & 0x80000000u) ? (u & 0x7FFFFFFFu) : ~u;
  return __uint_as_float(v);
}
__device__ __forceinline__ float fast_rcp(float x) {
#if __has_builtin(__builtin_amdgcn_rcpf)
  return __builtin_amdgcn_rcpf(x);            // v_rcp_f32
#else
  return 1.0f / x;
#endif
}
__device__ __forceinline__ float wave_sum(float v) {
  #pragma unroll
  for (int off = 16; off > 0; off >>= 1) v += __shfl_xor(v, off, 32);
  return v;
}

// ---------- utility kernels ----------
__global__ void k_zero_u32(unsigned* p, long long n) {
  long long i = (long long)blockIdx.x * blockDim.x + threadIdx.x;
  if (i < n) p[i] = 0u;
}
__global__ void k_cvt_bf16(const float* __restrict__ s, unsigned short* __restrict__ d, long long n) {
  long long i = (long long)blockIdx.x * blockDim.x + threadIdx.x;
  if (i < n) d[i] = f2b(s[i]);
}
// W[k, dout] fp32 row-major -> Wt[dout, k] bf16 (contiguous-K per output column)
__global__ void k_wprep(const float* __restrict__ W, unsigned short* __restrict__ Wt, int K, int Dout) {
  int i = blockIdx.x * blockDim.x + threadIdx.x;
  if (i < K * Dout) {
    int k = i / Dout, d = i % Dout;
    Wt[(size_t)d * K + k] = f2b(W[i]);
  }
}

// ---------- WMMA GEMM: out[M,Dout] = act(A[M,K](bf16) @ Bt[Dout,K]^T + bias) ----------
// 256-thread blocks = 8 waves; each wave owns one 16-row tile, caches the full
// A row-tile fragment in VGPRs (KS k-steps of 32), then sweeps all Dout/16
// column tiles with an uninterrupted WMMA chain per tile. All epilogue options
// are compile-time so the hot loop carries no branches.
// __launch_bounds__(256,1): allow the full VGPR budget so the KS=16 A-cache
// (128 VGPRs) stays resident instead of spilling to scratch — WMMA depth and
// load clauses hide latency here, not wave count.
// ACT: 0 none, 1 relu, 2 gelu(exact), 3 tanh, 4 sigmoid.  RESID: outF += acc first.
template<int KS, int ACT, bool RESID, bool HASBIAS, bool WF, bool WB>
__global__ __launch_bounds__(256, 1)
void k_gemm_bf16_wmma(const unsigned short* __restrict__ A,
                      const unsigned short* __restrict__ Bt,
                      const float* __restrict__ bias,
                      float* __restrict__ outF,
                      unsigned short* __restrict__ outB,
                      int M, int Dout)
{
  const int tileRow = blockIdx.x * 8 + (threadIdx.x >> 5);
  if (tileRow * 16 >= M) return;                 // whole wave exits together
  const int lane = threadIdx.x & 31;
  const int half = lane >> 4;                    // 0: lanes 0-15, 1: lanes 16-31
  const int mn   = lane & 15;                    // A row / B col within tile
  const int K    = KS * 32;
  const int ntiles = Dout >> 4;

  // cache A fragments for this row tile: lane holds K = s*32 + 16*half + {0..15}
  const unsigned short* arow = A + (size_t)(tileRow * 16 + mn) * K;
  v16bf afrag[KS];
  #pragma unroll
  for (int s = 0; s < KS; ++s) {
    afrag[s] = *(const v16bf*)(arow + 32 * s + 16 * half);
  }

  for (int t = 0; t < ntiles; ++t) {
    const int col = t * 16 + mn;
    const unsigned short* bcol = Bt + (size_t)col * K + 16 * half;
    v8f acc = {};
    #pragma unroll
    for (int s = 0; s < KS; ++s) {
      v16bf b = *(const v16bf*)(bcol + 32 * s);  // 16 contiguous K per lane
      acc = __builtin_amdgcn_wmma_f32_16x16x32_bf16(false, afrag[s], false, b,
                                                    (short)0, acc, false, false);
    }
    float bv = 0.0f;
    if constexpr (HASBIAS) bv = bias[col];
    #pragma unroll
    for (int j = 0; j < 8; ++j) {
      const size_t o = (size_t)(tileRow * 16 + j + 8 * half) * Dout + col;
      float v = acc[j] + bv;
      if constexpr (ACT == 1) v = fmaxf(v, 0.0f);
      if constexpr (ACT == 2) v = 0.5f * v * (1.0f + erff(v * 0.70710678118654752f));
      if constexpr (ACT == 3) {
#if __has_builtin(__builtin_amdgcn_tanhf)
        v = __builtin_amdgcn_tanhf(v);
#else
        v = tanhf(v);
#endif
      }
      if constexpr (ACT == 4) v = fast_rcp(1.0f + __expf(-v));
      if constexpr (RESID) v += outF[o];
      if constexpr (WF) outF[o] = v;
      if constexpr (WB) outB[o] = f2b(v);
    }
  }
}

// ---------- LayerNorm (one wave per node, 4 ch/lane) ----------
__global__ void k_layernorm(const float* __restrict__ hF,
                            const float* __restrict__ g, const float* __restrict__ b,
                            unsigned short* __restrict__ outB, int n)
{
  int wid  = (blockIdx.x * blockDim.x + threadIdx.x) >> 5;
  int lane = threadIdx.x & 31;
  if (wid >= n) return;
  const float4 v = *(const float4*)(hF + (size_t)wid * DD + lane * 4);
  float mu  = wave_sum(v.x + v.y + v.z + v.w) * (1.0f / DD);
  float dx = v.x - mu, dy = v.y - mu, dz = v.z - mu, dw = v.w - mu;
  float var = wave_sum(dx * dx + dy * dy + dz * dz + dw * dw) * (1.0f / DD);
  float inv = rsqrtf(var + 1e-5f);
  const float4 gg = *(const float4*)(g + lane * 4);
  const float4 bb = *(const float4*)(b + lane * 4);
  unsigned short* o = outB + (size_t)wid * DD + lane * 4;
  o[0] = f2b(dx * inv * gg.x + bb.x);
  o[1] = f2b(dy * inv * gg.y + bb.y);
  o[2] = f2b(dz * inv * gg.z + bb.z);
  o[3] = f2b(dw * inv * gg.w + bb.w);
}

// ---------- GATv2 edge pass 1: scores + segment max (one wave per edge) ----------
__global__ void k_edge_score(const int* __restrict__ ei,
                             const float* __restrict__ xl, const float* __restrict__ xr,
                             const float* __restrict__ att,
                             float* __restrict__ score, unsigned* __restrict__ smax)
{
  int wid  = (blockIdx.x * blockDim.x + threadIdx.x) >> 5;
  int lane = threadIdx.x & 31;
  const int ET = EDGES + NODES;
  if (wid >= ET) return;
  int s, d;
  if (wid < EDGES) { s = ei[wid]; d = ei[EDGES + wid]; } else { s = d = wid - EDGES; }
  const float4 a = *(const float4*)(xl + (size_t)s * DD + lane * 4);
  const float4 r = *(const float4*)(xr + (size_t)d * DD + lane * 4);
  const float4 w = *(const float4*)(att + lane * 4);
  float e0 = a.x + r.x; e0 = e0 > 0.f ? e0 : 0.2f * e0;
  float e1 = a.y + r.y; e1 = e1 > 0.f ? e1 : 0.2f * e1;
  float e2 = a.z + r.z; e2 = e2 > 0.f ? e2 : 0.2f * e2;
  float e3 = a.w + r.w; e3 = e3 > 0.f ? e3 : 0.2f * e3;
  float acc = wave_sum(e0 * w.x + e1 * w.y + e2 * w.z + e3 * w.w);
  if (lane == 0) { score[wid] = acc; atomicMax(smax + d, encf(acc)); }
}

// ---------- GATv2 edge pass 2: exp-weights + weighted feature accumulation ----------
__global__ void k_edge_accum(const int* __restrict__ ei,
                             const float* __restrict__ xl,
                             const float* __restrict__ score,
                             const unsigned* __restrict__ smax,
                             float* __restrict__ denom, float* __restrict__ accum)
{
  int wid  = (blockIdx.x * blockDim.x + threadIdx.x) >> 5;
  int lane = threadIdx.x & 31;
  const int ET = EDGES + NODES;
  if (wid >= ET) return;
  int s, d;
  if (wid < EDGES) { s = ei[wid]; d = ei[EDGES + wid]; } else { s = d = wid - EDGES; }
  const float ex = __expf(score[wid] - decf(smax[d]));
  if (lane == 0) unsafeAtomicAdd(denom + d, ex);
  const float4 a = *(const float4*)(xl + (size_t)s * DD + lane * 4);
  float* p = accum + (size_t)d * DD + lane * 4;
  unsafeAtomicAdd(p + 0, ex * a.x);
  unsafeAtomicAdd(p + 1, ex * a.y);
  unsafeAtomicAdd(p + 2, ex * a.z);
  unsafeAtomicAdd(p + 3, ex * a.w);
}

// ---------- node update: h += accum/denom + gat_bias ; refresh bf16 copy ----------
__global__ void k_node_update(const float* __restrict__ accum, const float* __restrict__ denom,
                              const float* __restrict__ gbias,
                              float* __restrict__ hF, unsigned short* __restrict__ hB)
{
  long long i = (long long)blockIdx.x * blockDim.x + threadIdx.x;
  if (i >= (long long)NODES * DD) return;
  int vtx = (int)(i >> 7), ch = (int)(i & 127);
  float v = hF[i] + accum[i] * fast_rcp(denom[vtx]) + gbias[ch];
  hF[i] = v;
  hB[i] = f2b(v);
}

// ---------- pooling pass 1: gated score per node + global max ----------
__global__ void k_pool_score(const float* __restrict__ a, const float* __restrict__ b,
                             const float* __restrict__ Wc, const float* __restrict__ bc,
                             float* __restrict__ score, unsigned* __restrict__ gmax)
{
  int wid  = (blockIdx.x * blockDim.x + threadIdx.x) >> 5;
  int lane = threadIdx.x & 31;
  if (wid >= NODES) return;
  const float4 av = *(const float4*)(a + (size_t)wid * DD + lane * 4);
  const float4 bv = *(const float4*)(b + (size_t)wid * DD + lane * 4);
  const float4 wv = *(const float4*)(Wc + lane * 4);
  float acc = wave_sum(av.x * bv.x * wv.x + av.y * bv.y * wv.y +
                       av.z * bv.z * wv.z + av.w * bv.w * wv.w);
  if (lane == 0) { float s = acc + bc[0]; score[wid] = s; atomicMax(gmax, encf(s)); }
}

// ---------- pooling pass 2: softmax-weighted sum of h ----------
__global__ void k_pool_accum(const float* __restrict__ score, const unsigned* __restrict__ gmax,
                             const float* __restrict__ hF,
                             float* __restrict__ gden, float* __restrict__ hpath)
{
  int wid  = (blockIdx.x * blockDim.x + threadIdx.x) >> 5;
  int lane = threadIdx.x & 31;
  if (wid >= NODES) return;
  const float w = __expf(score[wid] - decf(*gmax));
  if (lane == 0) unsafeAtomicAdd(gden, w);
  const float4 h = *(const float4*)(hF + (size_t)wid * DD + lane * 4);
  unsafeAtomicAdd(hpath + lane * 4 + 0, w * h.x);
  unsafeAtomicAdd(hpath + lane * 4 + 1, w * h.y);
  unsafeAtomicAdd(hpath + lane * 4 + 2, w * h.z);
  unsafeAtomicAdd(hpath + lane * 4 + 3, w * h.w);
}

// ---------- head: rho MLP + classifier + hazards/S/argmax ----------
__global__ void k_finalize(const float* __restrict__ hpath, const float* __restrict__ gden,
                           const float* __restrict__ rhoW, const float* __restrict__ rhob,
                           const float* __restrict__ clsW, const float* __restrict__ clsb,
                           float* __restrict__ out)
{
  __shared__ float hp[DD], hr[DD], lg[4];
  int t = threadIdx.x;
  hp[t] = hpath[t] / gden[0];
  __syncthreads();
  float acc = rhob[t];
  for (int k = 0; k < DD; ++k) acc += hp[k] * rhoW[k * DD + t];
  hr[t] = fmaxf(acc, 0.0f);
  __syncthreads();
  if (t < 4) {
    float l = clsb[t];
    for (int k = 0; k < DD; ++k) l += hr[k] * clsW[k * 4 + t];
    lg[t] = l;
  }
  __syncthreads();
  if (t == 0) {
    float S = 1.0f, best = lg[0]; int am = 0;
    for (int j = 0; j < 4; ++j) {
      float hz = 1.0f / (1.0f + __expf(-lg[j]));
      out[j] = hz;              // hazards
      S *= (1.0f - hz);
      out[4 + j] = S;           // survival
      out[9 + j] = lg[j];       // logits
      if (lg[j] > best) { best = lg[j]; am = j; }
    }
    out[8] = (float)am;         // Y_hat
  }
}

// ---------- host orchestration ----------
static inline int cdiv(long long a, long long b) { return (int)((a + b - 1) / b); }
static const dim3 GEMM_GRID((NODES / 16 + 7) / 8);
static const dim3 GEMM_BLK(256);

extern "C" void kernel_launch(void* const* d_in, const int* in_sizes, int n_in,
                              void* d_out, int out_size, void* d_ws, size_t ws_size,
                              hipStream_t stream) {
  (void)in_sizes; (void)n_in; (void)out_size; (void)ws_size;
  const float* x     = (const float*)d_in[0];
  const int*   ei    = (const int*)  d_in[1];
  const float* embW  = (const float*)d_in[2];
  const float* embB  = (const float*)d_in[3];
  const float* ln1g  = (const float*)d_in[4];
  const float* ln1b  = (const float*)d_in[5];
  const float* gWl   = (const float*)d_in[6];
  const float* gbl   = (const float*)d_in[7];
  const float* gWr   = (const float*)d_in[8];
  const float* gbr   = (const float*)d_in[9];
  const float* gatt  = (const float*)d_in[10];
  const float* gbias = (const float*)d_in[11];
  const float* w1[NBLK] = {(const float*)d_in[12], (const float*)d_in[14], (const float*)d_in[16]};
  const float* w2[NBLK] = {(const float*)d_in[13], (const float*)d_in[15], (const float*)d_in[17]};
  const float* Wa = (const float*)d_in[18]; const float* ba = (const float*)d_in[19];
  const float* Wb = (const float*)d_in[20]; const float* bb = (const float*)d_in[21];
  const float* Wc = (const float*)d_in[22]; const float* bc = (const float*)d_in[23];
  const float* rhoW = (const float*)d_in[24]; const float* rhob = (const float*)d_in[25];
  const float* clsW = (const float*)d_in[26]; const float* clsb = (const float*)d_in[27];
  float* out = (float*)d_out;

  const int ET = EDGES + NODES;
  // workspace carve-out
  char* base = (char*)d_ws; size_t cur = 0;
  auto alloc = [&](size_t bytes) -> void* {
    cur = (cur + 255) & ~(size_t)255; void* p = base + cur; cur += bytes; return p;
  };
  unsigned short* xB   = (unsigned short*)alloc((size_t)NODES * FIN * 2);
  float*          hF   = (float*)         alloc((size_t)NODES * DD * 4);
  unsigned short* hB   = (unsigned short*)alloc((size_t)NODES * DD * 2);
  unsigned short* hnB  = (unsigned short*)alloc((size_t)NODES * DD * 2);
  float*          xl   = (float*)         alloc((size_t)NODES * DD * 4);   // reused: attn a
  float*          xr   = (float*)         alloc((size_t)NODES * DD * 4);   // reused: attn b
  float*          accum= (float*)         alloc((size_t)NODES * DD * 4);
  unsigned short* hidB = (unsigned short*)alloc((size_t)NODES * 3 * DD * 2);
  float*          score= (float*)         alloc((size_t)ET * 4);
  unsigned*       smax = (unsigned*)      alloc((size_t)NODES * 4);
  float*          denom= (float*)         alloc((size_t)NODES * 4);
  unsigned*       pool = (unsigned*)      alloc((size_t)(2 + DD) * 4);     // gmax, gden, hpath[128]
  unsigned* gmax = pool;
  float*    gden = (float*)(pool + 1);
  float*    hpath= (float*)(pool + 2);
  unsigned short* embWt = (unsigned short*)alloc((size_t)FIN * DD * 2);
  unsigned short* WlT   = (unsigned short*)alloc((size_t)NBLK * DD * DD * 2);
  unsigned short* WrT   = (unsigned short*)alloc((size_t)NBLK * DD * DD * 2);
  unsigned short* w1T[NBLK], *w2T[NBLK];
  for (int i = 0; i < NBLK; ++i) {
    int h = DD * (i + 1);
    w1T[i] = (unsigned short*)alloc((size_t)DD * h * 2);
    w2T[i] = (unsigned short*)alloc((size_t)h * DD * 2);
  }
  unsigned short* WaT = (unsigned short*)alloc((size_t)DD * DD * 2);
  unsigned short* WbT = (unsigned short*)alloc((size_t)DD * DD * 2);

  const dim3 B256(256);
  const int waveGridN  = cdiv((long long)NODES * 32, 256);
  const int waveGridET = cdiv((long long)ET * 32, 256);

  // --- prep: bf16 activations, transposed bf16 weights ---
  k_cvt_bf16<<<cdiv((long long)NODES * FIN, 256), B256, 0, stream>>>(x, xB, (long long)NODES * FIN);
  k_wprep<<<cdiv(FIN * DD, 256), B256, 0, stream>>>(embW, embWt, FIN, DD);
  for (int i = 0; i < NBLK; ++i) {
    int h = DD * (i + 1);
    k_wprep<<<cdiv(DD * DD, 256), B256, 0, stream>>>(gWl + (size_t)i * DD * DD, WlT + (size_t)i * DD * DD, DD, DD);
    k_wprep<<<cdiv(DD * DD, 256), B256, 0, stream>>>(gWr + (size_t)i * DD * DD, WrT + (size_t)i * DD * DD, DD, DD);
    k_wprep<<<cdiv(DD * h, 256), B256, 0, stream>>>(w1[i], w1T[i], DD, h);
    k_wprep<<<cdiv(h * DD, 256), B256, 0, stream>>>(w2[i], w2T[i], h, DD);
  }
  k_wprep<<<cdiv(DD * DD, 256), B256, 0, stream>>>(Wa, WaT, DD, DD);
  k_wprep<<<cdiv(DD * DD, 256), B256, 0, stream>>>(Wb, WbT, DD, DD);

  // --- embedding: h = relu(x @ embW + b) ---  K=512, relu, bias, write f32+bf16
  k_gemm_bf16_wmma<16, 1, false, true, true, true>
      <<<GEMM_GRID, GEMM_BLK, 0, stream>>>(xB, embWt, embB, hF, hB, NODES, DD);

  // --- mixer blocks ---
  for (int i = 0; i < NBLK; ++i) {
    const int h = DD * (i + 1);
    k_layernorm<<<waveGridN, B256, 0, stream>>>(hF, ln1g + i * DD, ln1b + i * DD, hnB, NODES);
    // xl = hn @ Wl + bl ; xr = hn @ Wr + br   (K=128, no act, bias, f32 only)
    k_gemm_bf16_wmma<4, 0, false, true, true, false>
        <<<GEMM_GRID, GEMM_BLK, 0, stream>>>(hnB, WlT + (size_t)i * DD * DD, gbl + i * DD, xl, nullptr, NODES, DD);
    k_gemm_bf16_wmma<4, 0, false, true, true, false>
        <<<GEMM_GRID, GEMM_BLK, 0, stream>>>(hnB, WrT + (size_t)i * DD * DD, gbr + i * DD, xr, nullptr, NODES, DD);
    k_zero_u32<<<cdiv(NODES, 256), B256, 0, stream>>>(smax, NODES);
    k_zero_u32<<<cdiv(NODES, 256), B256, 0, stream>>>((unsigned*)denom, NODES);
    k_zero_u32<<<cdiv((long long)NODES * DD, 256), B256, 0, stream>>>((unsigned*)accum, (long long)NODES * DD);
    k_edge_score<<<waveGridET, B256, 0, stream>>>(ei, xl, xr, gatt + i * DD, score, smax);
    k_edge_accum<<<waveGridET, B256, 0, stream>>>(ei, xl, score, smax, denom, accum);
    k_node_update<<<cdiv((long long)NODES * DD, 256), B256, 0, stream>>>(accum, denom, gbias + i * DD, hF, hB);
    // MLP: h += gelu(h @ w1) @ w2
    k_gemm_bf16_wmma<4, 2, false, false, false, true>     // gelu, bf16 only
        <<<GEMM_GRID, GEMM_BLK, 0, stream>>>(hB, w1T[i], nullptr, nullptr, hidB, NODES, h);
    if (h == 128)
      k_gemm_bf16_wmma<4, 0, true, false, true, true>
          <<<GEMM_GRID, GEMM_BLK, 0, stream>>>(hidB, w2T[i], nullptr, hF, hB, NODES, DD);
    else if (h == 256)
      k_gemm_bf16_wmma<8, 0, true, false, true, true>
          <<<GEMM_GRID, GEMM_BLK, 0, stream>>>(hidB, w2T[i], nullptr, hF, hB, NODES, DD);
    else
      k_gemm_bf16_wmma<12, 0, true, false, true, true>
          <<<GEMM_GRID, GEMM_BLK, 0, stream>>>(hidB, w2T[i], nullptr, hF, hB, NODES, DD);
  }

  // --- gated attention pooling ---
  k_gemm_bf16_wmma<4, 3, false, true, true, false>        // tanh gate
      <<<GEMM_GRID, GEMM_BLK, 0, stream>>>(hB, WaT, ba, xl, nullptr, NODES, DD);
  k_gemm_bf16_wmma<4, 4, false, true, true, false>        // sigmoid gate
      <<<GEMM_GRID, GEMM_BLK, 0, stream>>>(hB, WbT, bb, xr, nullptr, NODES, DD);
  k_zero_u32<<<1, B256, 0, stream>>>(pool, 2 + DD);
  k_pool_score<<<waveGridN, B256, 0, stream>>>(xl, xr, Wc, bc, score, gmax);
  k_pool_accum<<<waveGridN, B256, 0, stream>>>(score, gmax, hF, gden, hpath);

  // --- head ---
  k_finalize<<<1, dim3(DD), 0, stream>>>(hpath, gden, rhoW, rhob, clsW, clsb, out);
}
